// DeepKalmanFilter_65455301591444
// MI455X (gfx1250) — matrix operand order repaired and checked by reading, not compiled, not run on hardware
//
#include <hip/hip_runtime.h>
#include <hip/hip_bf16.h>
#include <math.h>
#include <stdint.h>

#define B_  512
#define T_  256
#define DIN 32
#define ZD  64
#define ED  256
#define TR  256
#define RD  128
#define OC  288   // 32 + 4*64 output columns

typedef __bf16 bf16;
typedef __attribute__((ext_vector_type(16))) __bf16 v16bf;
typedef __attribute__((ext_vector_type(8)))  float  v8f;

static __device__ inline v8f wmma_bf(v16bf a, v16bf b, v8f c) {
  // D = A(16x32 bf16) * B(32x16 bf16) + C(16x16 f32)
  return __builtin_amdgcn_wmma_f32_16x16x32_bf16(false, a, false, b, (short)0, c,
                                                 false, false);
}

static __device__ inline v8f splat8(float v) {
  v8f c;
#pragma unroll
  for (int i = 0; i < 8; ++i) c[i] = v;
  return c;
}

static __device__ inline float softplusf(float x) {
  return (x > 20.f) ? x : log1pf(__expf(x));
}

// A-operand fragment from row-major [16][ldk] bf16 tile (LDS or global).
// 16-bit A layout: lanes 0-15 -> M=lane, K in {kb..kb+7, kb+16..kb+23};
// lanes 16-31 -> M=lane-16, K shifted by 8. Two contiguous 16B chunks/lane.
static __device__ inline v16bf loadA(const bf16* base, int ldk, int lane, int kbase) {
  const int m  = lane & 15;
  const int h8 = (lane >> 4) * 8;
  const bf16* p = base + m * ldk + kbase + h8;
  v16bf a;
#pragma unroll
  for (int i = 0; i < 8; ++i) { a[i] = p[i]; a[8 + i] = p[16 + i]; }
  return a;
}

// Same A fragment but sourced from f32 global data (converted on the fly).
static __device__ inline v16bf loadA_f32(const float* base, int ldk, int lane, int kbase) {
  const int m  = lane & 15;
  const int h8 = (lane >> 4) * 8;
  const float* p = base + m * ldk + kbase + h8;
  v16bf a;
#pragma unroll
  for (int i = 0; i < 8; ++i) { a[i] = (bf16)p[i]; a[8 + i] = (bf16)p[16 + i]; }
  return a;
}

// B-operand fragment from [N][K] row-major weight (out_features x in_features).
// Lane L holds column n = n0 + (L&15), K in {kb+8h..+7, kb+16+8h..+7}, h = L>>4:
// two 16-byte reads (matches the A layout).
static __device__ inline v16bf loadB(const bf16* W, int ldk, int lane, int n0, int kbase) {
  const int n = n0 + (lane & 15);
  const int k = kbase + ((lane >> 4) * 8);
  const bf16* p = W + n * ldk + k;
  v16bf b;
#pragma unroll
  for (int i = 0; i < 8; ++i) { b[i] = p[i]; b[8 + i] = p[16 + i]; }
  return b;
}

// Async DMA of a weight segment into LDS (ASYNCcnt-tracked, no VGPR staging).
// bytes must be a multiple of 4096 so every lane of every wave participates
// each iteration (EXEC stays all-ones, no divergence).
static __device__ inline void async_copy_seg(const bf16* __restrict__ src,
                                             bf16* dst, int bytes, int tid) {
  const char* g = (const char*)src;
  char* l = (char*)dst;
  for (int i = tid * 16; i < bytes; i += 256 * 16) {
    uint32_t lds_off = (uint32_t)(uintptr_t)(l + i);   // low 32 bits = LDS addr
    const void* gp = (const void*)(g + i);
    asm volatile("global_load_async_to_lds_b128 %0, %1, off"
                 :: "v"(lds_off), "v"(gp)
                 : "memory");
  }
}

// ---------------------------------------------------------------------------
// fp32 -> bf16 weight conversion (once per launch, stream-ordered)
// ---------------------------------------------------------------------------
__global__ void cvt_bf16(const float* __restrict__ s, bf16* __restrict__ d, int n) {
  int i = blockIdx.x * 256 + threadIdx.x;
  if (i < n) d[i] = (bf16)s[i];
}

// ---------------------------------------------------------------------------
// Scan kernel: fused RNN + combiner + gated transition, sequential over T.
// One workgroup (8 waves) per 16-row batch tile; waves split the N dimension.
// Hot recurrent weights live in LDS (filled by async-to-LDS DMA); the wide
// transition weights (Wg1/Wg2/Wp1/Wp2) stay in global/L2 where their stages
// have enough independent WMMAs to hide the latency.
// ---------------------------------------------------------------------------
// LDS weight arena offsets (bf16 elements)
#define WOFF_IH  0                   // W_ih  [128][32]  -> 4096
#define WOFF_HH  4096                // W_hh  [128][128] -> 16384
#define WOFF_ZH  20480               // Wc_zh [128][64]  -> 8192
#define WOFF_LOC 28672               // Wc_loc[64][128]  -> 8192
#define WOFF_SC  36864               // Wc_sc [64][128]  -> 8192
#define WOFF_ZL  45056               // Wt_zl [64][64]   -> 4096
#define WOFF_SIG 49152               // Wt_sig[64][64]   -> 4096
#define WTOT     53248               // 104 KB

struct ScanArgs {
  const float* seq; const float* eps;
  const float *b_ih, *b_hh, *bc_zh, *bc_loc, *bc_sc;
  const float *bt_g1, *bt_g2, *bt_p1, *bt_p2, *bt_zl, *bt_sig;
  const float *h0, *z_q0;
  const bf16 *Wih, *Whh, *Wzh, *Wloc, *Wsc, *Wg1, *Wg2, *Wp1, *Wp2, *Wzl, *Wsig;
  bf16* z_all;
  float* out;
};

__global__ __launch_bounds__(256) void dkf_scan(ScanArgs A) {
  __shared__ __align__(16) bf16  sh_w [WTOT];     // resident weights (104 KB)
  __shared__ __align__(16) bf16  sh_h [16 * RD];
  __shared__ __align__(16) bf16  sh_z [16 * ZD];
  __shared__ __align__(16) bf16  sh_qh[16 * RD];
  __shared__ __align__(16) bf16  sh_g1[16 * TR];
  __shared__ __align__(16) bf16  sh_ph[16 * TR];
  __shared__ __align__(16) bf16  sh_rp[16 * ZD];
  __shared__ float sh_qloc[16 * ZD];
  __shared__ float sh_qsc [16 * ZD];
  __shared__ float sh_gate[16 * ZD];
  __shared__ float sh_prop[16 * ZD];
  __shared__ float sh_zl  [16 * ZD];

  const int tid  = threadIdx.x;
  const int w    = tid >> 5;          // wave id 0..7 (wave32)
  const int lane = tid & 31;
  const int m0   = blockIdx.x * 16;   // batch tile start
  const int cm   = (lane >> 4) * 8;   // C-fragment row base
  const int cn   = lane & 15;         // C-fragment column within tile

  // --- async DMA of hot weights into LDS (26 x b128 per lane, ASYNCcnt<=26)
  async_copy_seg(A.Wih,  sh_w + WOFF_IH,  RD * DIN * 2, tid);
  async_copy_seg(A.Whh,  sh_w + WOFF_HH,  RD * RD  * 2, tid);
  async_copy_seg(A.Wzh,  sh_w + WOFF_ZH,  RD * ZD  * 2, tid);
  async_copy_seg(A.Wloc, sh_w + WOFF_LOC, ZD * RD  * 2, tid);
  async_copy_seg(A.Wsc,  sh_w + WOFF_SC,  ZD * RD  * 2, tid);
  async_copy_seg(A.Wzl,  sh_w + WOFF_ZL,  ZD * ZD  * 2, tid);
  async_copy_seg(A.Wsig, sh_w + WOFF_SIG, ZD * ZD  * 2, tid);

  for (int i = tid; i < 16 * RD; i += 256) sh_h[i] = (bf16)A.h0[i & (RD - 1)];
  for (int i = tid; i < 16 * ZD; i += 256) sh_z[i] = (bf16)A.z_q0[i & (ZD - 1)];
  asm volatile("s_wait_asynccnt 0" ::: "memory");
  __syncthreads();

  for (int t = 0; t < T_; ++t) {
    // --- snapshot recurrent state as A fragments (registers) ---
    v16bf zA0 = loadA(sh_z, ZD, lane, 0);
    v16bf zA1 = loadA(sh_z, ZD, lane, 32);
    v16bf hA0 = loadA(sh_h, RD, lane, 0);
    v16bf hA1 = loadA(sh_h, RD, lane, 32);
    v16bf hA2 = loadA(sh_h, RD, lane, 64);
    v16bf hA3 = loadA(sh_h, RD, lane, 96);
    const float* sbase = A.seq + ((long)m0 * T_ + t) * DIN;
    v16bf sA = loadA_f32(sbase, T_ * DIN, lane, 0);
    __syncthreads();   // all state reads done before overwrites

    // --- stage 1: h_t = relu(seq@Wih^T + h@Whh^T + b) ; N=128 -> ntile = w ---
    {
      const int n0 = w * 16;
      v8f c = splat8(A.b_ih[n0 + cn] + A.b_hh[n0 + cn]);
      c = wmma_bf(sA,  loadB(sh_w + WOFF_IH, DIN, lane, n0, 0),  c);
      c = wmma_bf(hA0, loadB(sh_w + WOFF_HH, RD,  lane, n0, 0),  c);
      c = wmma_bf(hA1, loadB(sh_w + WOFF_HH, RD,  lane, n0, 32), c);
      c = wmma_bf(hA2, loadB(sh_w + WOFF_HH, RD,  lane, n0, 64), c);
      c = wmma_bf(hA3, loadB(sh_w + WOFF_HH, RD,  lane, n0, 96), c);
#pragma unroll
      for (int r = 0; r < 8; ++r)
        sh_h[(cm + r) * RD + n0 + cn] = (bf16)fmaxf(c[r], 0.f);
    }
    __syncthreads();

    // --- stage 2: q_h = 0.5*(tanh(z@Wzh^T + b) + h_t) ; N=128 -> ntile = w ---
    {
      const int n0 = w * 16;
      v8f c = splat8(A.bc_zh[n0 + cn]);
      c = wmma_bf(zA0, loadB(sh_w + WOFF_ZH, ZD, lane, n0, 0),  c);
      c = wmma_bf(zA1, loadB(sh_w + WOFF_ZH, ZD, lane, n0, 32), c);
#pragma unroll
      for (int r = 0; r < 8; ++r) {
        const int idx = (cm + r) * RD + n0 + cn;
        float qh = 0.5f * (tanhf(c[r]) + (float)sh_h[idx]);
        sh_qh[idx] = (bf16)qh;
      }
    }
    __syncthreads();

    // --- stage 3: q_loc (waves 0-3) / q_scale (waves 4-7), N=64 each ---
    {
      v16bf qA0 = loadA(sh_qh, RD, lane, 0);
      v16bf qA1 = loadA(sh_qh, RD, lane, 32);
      v16bf qA2 = loadA(sh_qh, RD, lane, 64);
      v16bf qA3 = loadA(sh_qh, RD, lane, 96);
      const int  n0    = (w & 3) * 16;
      const bool isLoc = (w < 4);                 // wave-uniform
      const float* bias = isLoc ? A.bc_loc : A.bc_sc;
      const bf16*  W    = isLoc ? (sh_w + WOFF_LOC) : (sh_w + WOFF_SC);
      v8f c = splat8(bias[n0 + cn]);
      c = wmma_bf(qA0, loadB(W, RD, lane, n0, 0),  c);
      c = wmma_bf(qA1, loadB(W, RD, lane, n0, 32), c);
      c = wmma_bf(qA2, loadB(W, RD, lane, n0, 64), c);
      c = wmma_bf(qA3, loadB(W, RD, lane, n0, 96), c);
      if (isLoc) {
#pragma unroll
        for (int r = 0; r < 8; ++r) {
          sh_qloc[(cm + r) * ZD + n0 + cn] = c[r];
          long o = ((long)(m0 + cm + r) * T_ + t) * OC;
          A.out[o + 32 + n0 + cn] = c[r];
        }
      } else {
#pragma unroll
        for (int r = 0; r < 8; ++r) {
          float s = softplusf(c[r]);
          sh_qsc[(cm + r) * ZD + n0 + cn] = s;
          long o = ((long)(m0 + cm + r) * T_ + t) * OC;
          A.out[o + 96 + n0 + cn] = s;
        }
      }
    }
    __syncthreads();

    // --- stage 4: g1 = relu(z@Wg1^T+b), ph = relu(z@Wp1^T+b) ; N=256 each ---
#pragma unroll
    for (int j = 0; j < 2; ++j) {
      const int n0 = (w * 2 + j) * 16;
      v8f c = splat8(A.bt_g1[n0 + cn]);
      c = wmma_bf(zA0, loadB(A.Wg1, ZD, lane, n0, 0),  c);
      c = wmma_bf(zA1, loadB(A.Wg1, ZD, lane, n0, 32), c);
      v8f d = splat8(A.bt_p1[n0 + cn]);
      d = wmma_bf(zA0, loadB(A.Wp1, ZD, lane, n0, 0),  d);
      d = wmma_bf(zA1, loadB(A.Wp1, ZD, lane, n0, 32), d);
#pragma unroll
      for (int r = 0; r < 8; ++r) {
        sh_g1[(cm + r) * TR + n0 + cn] = (bf16)fmaxf(c[r], 0.f);
        sh_ph[(cm + r) * TR + n0 + cn] = (bf16)fmaxf(d[r], 0.f);
      }
    }
    __syncthreads();

    // --- stage 5: gate + zl (waves 0-3) / prop (waves 4-7) ---
    if (w < 4) {
      const int n0 = w * 16;
      v8f c = splat8(A.bt_g2[n0 + cn]);
#pragma unroll
      for (int kt = 0; kt < 8; ++kt)
        c = wmma_bf(loadA(sh_g1, TR, lane, kt * 32),
                    loadB(A.Wg2, TR, lane, n0, kt * 32), c);
      v8f d = splat8(A.bt_zl[n0 + cn]);
      d = wmma_bf(zA0, loadB(sh_w + WOFF_ZL, ZD, lane, n0, 0),  d);
      d = wmma_bf(zA1, loadB(sh_w + WOFF_ZL, ZD, lane, n0, 32), d);
#pragma unroll
      for (int r = 0; r < 8; ++r) {
        const int idx = (cm + r) * ZD + n0 + cn;
        sh_gate[idx] = 1.f / (1.f + __expf(-c[r]));
        sh_zl[idx]   = d[r];
      }
    } else {
      const int n0 = (w - 4) * 16;
      v8f c = splat8(A.bt_p2[n0 + cn]);
#pragma unroll
      for (int kt = 0; kt < 8; ++kt)
        c = wmma_bf(loadA(sh_ph, TR, lane, kt * 32),
                    loadB(A.Wp2, TR, lane, n0, kt * 32), c);
#pragma unroll
      for (int r = 0; r < 8; ++r) {
        const int idx = (cm + r) * ZD + n0 + cn;
        sh_prop[idx] = c[r];
        sh_rp[idx]   = (bf16)fmaxf(c[r], 0.f);
      }
    }
    __syncthreads();

    // --- stage 5b: p_loc blend (waves 0-3) / p_scale GEMM (waves 4-7) ---
    if (w < 4) {
      const int n0 = w * 16;
#pragma unroll
      for (int r = 0; r < 8; ++r) {
        const int mi = cm + r, ni = n0 + cn;
        float g  = sh_gate[mi * ZD + ni];
        float pl = (1.f - g) * sh_zl[mi * ZD + ni] + g * sh_prop[mi * ZD + ni];
        long o = ((long)(m0 + mi) * T_ + t) * OC;
        A.out[o + 160 + ni] = pl;
      }
    } else {
      const int n0 = (w - 4) * 16;
      v16bf rA0 = loadA(sh_rp, ZD, lane, 0);
      v16bf rA1 = loadA(sh_rp, ZD, lane, 32);
      v8f c = splat8(A.bt_sig[n0 + cn]);
      c = wmma_bf(rA0, loadB(sh_w + WOFF_SIG, ZD, lane, n0, 0),  c);
      c = wmma_bf(rA1, loadB(sh_w + WOFF_SIG, ZD, lane, n0, 32), c);
#pragma unroll
      for (int r = 0; r < 8; ++r) {
        long o = ((long)(m0 + cm + r) * T_ + t) * OC;
        A.out[o + 224 + n0 + cn] = softplusf(c[r]);
      }
    }

    // --- stage 6: z_t = q_loc + q_scale * eps (all 256 threads) ---
    for (int i = tid; i < 16 * ZD; i += 256) {
      const int mi = i >> 6, ni = i & 63;
      long row = (long)(m0 + mi) * T_ + t;
      float zv = sh_qloc[i] + sh_qsc[i] * A.eps[row * ZD + ni];
      sh_z[i] = (bf16)zv;
      A.z_all[row * ZD + ni] = (bf16)zv;
    }
    __syncthreads();
  }
}

// ---------------------------------------------------------------------------
// Emitter kernel: fully parallel over B*T rows (FLOP-heavy 43% of the model).
// ---------------------------------------------------------------------------
struct EmitArgs {
  const bf16* z_all;
  const bf16 *We1, *We2, *We3;
  const float *be1, *be2, *be3;
  float* out;
};

__global__ __launch_bounds__(256) void dkf_emit(EmitArgs A) {
  __shared__ __align__(16) bf16 sh_e1[16 * ED];
  __shared__ __align__(16) bf16 sh_e2[16 * ED];
  const int tid = threadIdx.x, w = tid >> 5, lane = tid & 31;
  const int cm = (lane >> 4) * 8, cn = lane & 15;
  const long r0 = (long)blockIdx.x * 16;   // flat row = b*T + t

  const bf16* zb = A.z_all + r0 * ZD;
  v16bf zA0 = loadA(zb, ZD, lane, 0);
  v16bf zA1 = loadA(zb, ZD, lane, 32);

  // e1 = relu(z @ We1^T + be1), N=256 -> 2 ntiles per wave
#pragma unroll
  for (int j = 0; j < 2; ++j) {
    const int n0 = (w * 2 + j) * 16;
    v8f c = splat8(A.be1[n0 + cn]);
    c = wmma_bf(zA0, loadB(A.We1, ZD, lane, n0, 0),  c);
    c = wmma_bf(zA1, loadB(A.We1, ZD, lane, n0, 32), c);
#pragma unroll
    for (int r = 0; r < 8; ++r)
      sh_e1[(cm + r) * ED + n0 + cn] = (bf16)fmaxf(c[r], 0.f);
  }
  __syncthreads();

  // e2 = relu(e1 @ We2^T + be2), 256x256 GEMM, K=256 -> 8 WMMA per tile
  v16bf eA[8];
#pragma unroll
  for (int kt = 0; kt < 8; ++kt) eA[kt] = loadA(sh_e1, ED, lane, kt * 32);
#pragma unroll
  for (int j = 0; j < 2; ++j) {
    const int n0 = (w * 2 + j) * 16;
    v8f c = splat8(A.be2[n0 + cn]);
#pragma unroll
    for (int kt = 0; kt < 8; ++kt)
      c = wmma_bf(eA[kt], loadB(A.We2, ED, lane, n0, kt * 32), c);
#pragma unroll
    for (int r = 0; r < 8; ++r)
      sh_e2[(cm + r) * ED + n0 + cn] = (bf16)fmaxf(c[r], 0.f);
  }
  __syncthreads();

  // mu = e2 @ We3^T + be3, N=32 -> waves 0,1
  if (w < 2) {
    const int n0 = w * 16;
    v8f c = splat8(A.be3[n0 + cn]);
#pragma unroll
    for (int kt = 0; kt < 8; ++kt)
      c = wmma_bf(loadA(sh_e2, ED, lane, kt * 32),
                  loadB(A.We3, ED, lane, n0, kt * 32), c);
#pragma unroll
    for (int r = 0; r < 8; ++r)
      A.out[(r0 + cm + r) * OC + n0 + cn] = c[r];
  }
}

// ---------------------------------------------------------------------------
extern "C" void kernel_launch(void* const* d_in, const int* in_sizes, int n_in,
                              void* d_out, int out_size, void* d_ws, size_t ws_size,
                              hipStream_t stream) {
  (void)in_sizes; (void)n_in; (void)out_size; (void)ws_size;
  const float* seq   = (const float*)d_in[0];
  const float* eps   = (const float*)d_in[1];
  const float* W_ih  = (const float*)d_in[2];
  const float* b_ih  = (const float*)d_in[3];
  const float* W_hh  = (const float*)d_in[4];
  const float* b_hh  = (const float*)d_in[5];
  const float* h0    = (const float*)d_in[6];
  const float* z_q0  = (const float*)d_in[7];
  const float* Wc_zh = (const float*)d_in[8];
  const float* bc_zh = (const float*)d_in[9];
  const float* Wc_lo = (const float*)d_in[10];
  const float* bc_lo = (const float*)d_in[11];
  const float* Wc_sc = (const float*)d_in[12];
  const float* bc_sc = (const float*)d_in[13];
  const float* Wt_g1 = (const float*)d_in[14];
  const float* bt_g1 = (const float*)d_in[15];
  const float* Wt_g2 = (const float*)d_in[16];
  const float* bt_g2 = (const float*)d_in[17];
  const float* Wt_p1 = (const float*)d_in[18];
  const float* bt_p1 = (const float*)d_in[19];
  const float* Wt_p2 = (const float*)d_in[20];
  const float* bt_p2 = (const float*)d_in[21];
  const float* Wt_zl = (const float*)d_in[22];
  const float* bt_zl = (const float*)d_in[23];
  const float* Wt_sg = (const float*)d_in[24];
  const float* bt_sg = (const float*)d_in[25];
  const float* We1   = (const float*)d_in[26];
  const float* be1   = (const float*)d_in[27];
  const float* We2   = (const float*)d_in[28];
  const float* be2   = (const float*)d_in[29];
  const float* We3   = (const float*)d_in[30];
  const float* be3   = (const float*)d_in[31];

  char* ws = (char*)d_ws;
  size_t off = 0;
  auto alloc = [&](size_t bytes) -> char* {
    char* p = ws + off;
    off = (off + bytes + 255) & ~(size_t)255;
    return p;
  };

  bf16* z_all = (bf16*)alloc((size_t)B_ * T_ * ZD * sizeof(bf16));
  bf16* cWih  = (bf16*)alloc((size_t)RD * DIN * 2);
  bf16* cWhh  = (bf16*)alloc((size_t)RD * RD  * 2);
  bf16* cWzh  = (bf16*)alloc((size_t)RD * ZD  * 2);
  bf16* cWlo  = (bf16*)alloc((size_t)ZD * RD  * 2);
  bf16* cWsc  = (bf16*)alloc((size_t)ZD * RD  * 2);
  bf16* cWg1  = (bf16*)alloc((size_t)TR * ZD  * 2);
  bf16* cWg2  = (bf16*)alloc((size_t)ZD * TR  * 2);
  bf16* cWp1  = (bf16*)alloc((size_t)TR * ZD  * 2);
  bf16* cWp2  = (bf16*)alloc((size_t)ZD * TR  * 2);
  bf16* cWzl  = (bf16*)alloc((size_t)ZD * ZD  * 2);
  bf16* cWsg  = (bf16*)alloc((size_t)ZD * ZD  * 2);
  bf16* cWe1  = (bf16*)alloc((size_t)ED * ZD  * 2);
  bf16* cWe2  = (bf16*)alloc((size_t)ED * ED  * 2);
  bf16* cWe3  = (bf16*)alloc((size_t)DIN * ED * 2);

  auto cv = [&](const float* s, bf16* d, int n) {
    cvt_bf16<<<(n + 255) / 256, 256, 0, stream>>>(s, d, n);
  };
  cv(W_ih,  cWih, RD * DIN);
  cv(W_hh,  cWhh, RD * RD);
  cv(Wc_zh, cWzh, RD * ZD);
  cv(Wc_lo, cWlo, ZD * RD);
  cv(Wc_sc, cWsc, ZD * RD);
  cv(Wt_g1, cWg1, TR * ZD);
  cv(Wt_g2, cWg2, ZD * TR);
  cv(Wt_p1, cWp1, TR * ZD);
  cv(Wt_p2, cWp2, ZD * TR);
  cv(Wt_zl, cWzl, ZD * ZD);
  cv(Wt_sg, cWsg, ZD * ZD);
  cv(We1,   cWe1, ED * ZD);
  cv(We2,   cWe2, ED * ED);
  cv(We3,   cWe3, DIN * ED);

  ScanArgs SA;
  SA.seq = seq; SA.eps = eps;
  SA.b_ih = b_ih; SA.b_hh = b_hh; SA.bc_zh = bc_zh; SA.bc_loc = bc_lo; SA.bc_sc = bc_sc;
  SA.bt_g1 = bt_g1; SA.bt_g2 = bt_g2; SA.bt_p1 = bt_p1; SA.bt_p2 = bt_p2;
  SA.bt_zl = bt_zl; SA.bt_sig = bt_sg;
  SA.h0 = h0; SA.z_q0 = z_q0;
  SA.Wih = cWih; SA.Whh = cWhh; SA.Wzh = cWzh; SA.Wloc = cWlo; SA.Wsc = cWsc;
  SA.Wg1 = cWg1; SA.Wg2 = cWg2; SA.Wp1 = cWp1; SA.Wp2 = cWp2;
  SA.Wzl = cWzl; SA.Wsig = cWsg;
  SA.z_all = z_all; SA.out = (float*)d_out;
  dkf_scan<<<B_ / 16, 256, 0, stream>>>(SA);

  EmitArgs EA;
  EA.z_all = z_all;
  EA.We1 = cWe1; EA.We2 = cWe2; EA.We3 = cWe3;
  EA.be1 = be1; EA.be2 = be2; EA.be3 = be3;
  EA.out = (float*)d_out;
  dkf_emit<<<(B_ * T_) / 16, 256, 0, stream>>>(EA);
}
